// CausalSelfAttentionKV_85899345920477
// MI455X (gfx1250) — matrix operand order repaired and checked by reading, not compiled, not run on hardware
//
#include <hip/hip_runtime.h>
#include <hip/hip_bf16.h>
#include <stdint.h>

#define B_   4
#define T_   2048
#define C_   2048
#define NH_  16
#define HD_  128

typedef __bf16 v16bf __attribute__((ext_vector_type(16)));
typedef float  v8f   __attribute__((ext_vector_type(8)));
typedef uint32_t u32x4 __attribute__((ext_vector_type(4)));
typedef int32_t  i32x8 __attribute__((ext_vector_type(8)));
typedef int32_t  i32x4 __attribute__((ext_vector_type(4)));
typedef int v4i_vs __attribute__((vector_size(16)));  // matches builtin param type

union FragBF { uint4 u4[2]; v16bf v; };

#ifndef __has_builtin
#define __has_builtin(x) 0
#endif
#define HAS_ASYNC_LDS   __has_builtin(__builtin_amdgcn_global_load_async_to_lds_b128)
#define HAS_WAIT_ASYNC  __has_builtin(__builtin_amdgcn_s_wait_asynccnt)
#define HAS_TDM         __has_builtin(__builtin_amdgcn_tensor_load_to_lds)
#define HAS_WAIT_TENSOR __has_builtin(__builtin_amdgcn_s_wait_tensorcnt)

static __device__ __forceinline__ unsigned short f2bf(float f) {
  union { float f; uint32_t u; } v; v.f = f;
  uint32_t r = v.u + 0x7FFFu + ((v.u >> 16) & 1u);  // round-to-nearest-even
  return (unsigned short)(r >> 16);
}

static __device__ __forceinline__ v8f zero8() {
  v8f z = {0.f, 0.f, 0.f, 0.f, 0.f, 0.f, 0.f, 0.f};
  return z;
}

static __device__ __forceinline__ v8f wmma_bf16(v16bf a, v16bf b, v8f c) {
  return __builtin_amdgcn_wmma_f32_16x16x32_bf16(false, a, false, b, (short)0, c,
                                                 false, false);
}

// 16B global -> LDS copy; async direct-to-LDS when available (ASYNCcnt path)
static __device__ __forceinline__ void copy16_g2l(const unsigned short* g,
                                                  unsigned short* l) {
#if HAS_ASYNC_LDS
  __builtin_amdgcn_global_load_async_to_lds_b128(
      (__attribute__((address_space(1))) v4i_vs*)(uintptr_t)g,
      (__attribute__((address_space(3))) v4i_vs*)(uint32_t)(uintptr_t)l,
      0, 0);
#else
  *(uint4*)l = *(const uint4*)g;
#endif
}

static __device__ __forceinline__ void wait_async_lds() {
#if HAS_ASYNC_LDS && HAS_WAIT_ASYNC
  __builtin_amdgcn_s_wait_asynccnt(0);
#endif
}

// TDM: DMA a 64x128 bf16 tile (row stride 128 elems in memory) into LDS with
// 16B padding after each 256B row (-> LDS row stride 272B = 136 elems).
static __device__ __forceinline__ void tdm_load_k_tile(const unsigned short* gsrc,
                                                       unsigned short* ldst) {
#if HAS_TDM
  const uint64_t ga = (uint64_t)(uintptr_t)gsrc;
  const uint32_t la = (uint32_t)(uintptr_t)ldst;
  u32x4 g0;
  g0[0] = 1u;                                            // count=1 (valid), user mode
  g0[1] = la;                                            // lds_addr
  g0[2] = (uint32_t)ga;                                  // global_addr[31:0]
  g0[3] = (uint32_t)((ga >> 32) & 0x01FFFFFFu) | (2u << 30);  // addr[56:32] | type=2
  i32x8 g1;
  g1[0] = (1 << 16) |      // data_size = 2 bytes
          (1 << 20) |      // pad_enable
          (5 << 22) |      // pad_interval: 64 DWORDs = 256B (one row)
          (3 << 25);       // pad_amount: 4 DWORDs = 16B
  g1[1] = (int32_t)(128u << 16);   // tensor_dim0 = 128 (bits 79:48, low half)
  g1[2] = (int32_t)(2048u << 16);  // tensor_dim0 hi = 0 | tensor_dim1 = 2048
  g1[3] = (int32_t)(128u << 16);   // tensor_dim1 hi = 0 | tile_dim0 = 128
  g1[4] = 64;                      // tile_dim1 = 64, tile_dim2 = 0
  g1[5] = 128;                     // tensor_dim0_stride = 128
  g1[6] = 0;
  g1[7] = 0;
  i32x4 gz4 = {0, 0, 0, 0};
  i32x8 gz8 = {0, 0, 0, 0, 0, 0, 0, 0};
  __builtin_amdgcn_tensor_load_to_lds(g0, g1, gz4, gz4, gz8, 0);
#else
  (void)gsrc; (void)ldst;
#endif
}

static __device__ __forceinline__ void wait_tensor() {
#if HAS_TDM && HAS_WAIT_TENSOR
  __builtin_amdgcn_s_wait_tensorcnt(0);
#endif
}

// A fragment (16x32, row-major source [row][k]):
static __device__ __forceinline__ v16bf frag_a(const unsigned short* lds, int stride,
                                               int row0, int kbase, int lane) {
  FragBF f;
  const int m  = lane & 15;
  const int k8 = (lane >> 4) * 8;
  const unsigned short* p = lds + (row0 + m) * stride + kbase;
  f.u4[0] = *(const uint4*)(p + k8);
  f.u4[1] = *(const uint4*)(p + 16 + k8);
  return f.v;
}

// B fragment (32x16, source stored [n][k] row-major over k):
static __device__ __forceinline__ v16bf frag_b(const unsigned short* lds, int stride,
                                               int col0, int kbase, int lane) {
  FragBF f;
  const int n   = lane & 15;
  const int k8  = (lane >> 4) * 8;
  const unsigned short* p = lds + (col0 + n) * stride + kbase;
  f.u4[0] = *(const uint4*)(p + k8);
  f.u4[1] = *(const uint4*)(p + 16 + k8);
  return f.v;
}

// ---------------------------------------------------------------------------
// Kernel 1: qkv = x @ Wqkv^T + bqkv ; scatter into head-major q/k/v
// ---------------------------------------------------------------------------
__launch_bounds__(256)
__global__ void qkv_gemm_kernel(const float* __restrict__ x,
                                const float* __restrict__ Wqkv,
                                const float* __restrict__ bqkv,
                                float* __restrict__ outk,
                                float* __restrict__ outv,
                                unsigned short* __restrict__ qb,
                                unsigned short* __restrict__ kbuf,
                                unsigned short* __restrict__ vbuf) {
  constexpr int LDSS = 40;
  __shared__ __align__(16) unsigned short As[128 * LDSS];
  __shared__ __align__(16) unsigned short Bs[128 * LDSS];
  const int tid  = threadIdx.x;
  const int lane = tid & 31;
  const int wave = tid >> 5;
  const int m0 = blockIdx.x * 128;
  const int n0 = blockIdx.y * 128;
  const int wm = (wave >> 2) * 64;
  const int wn = (wave & 3) * 32;

  v8f acc[4][2];
#pragma unroll
  for (int i = 0; i < 4; ++i)
#pragma unroll
    for (int j = 0; j < 2; ++j) acc[i][j] = zero8();

  for (int k0 = 0; k0 < C_; k0 += 32) {
    if (k0 + 32 < C_)
      __builtin_prefetch(x + (size_t)(m0 + (tid >> 3)) * C_ + k0 + 32, 0, 1);
#pragma unroll
    for (int i = 0; i < 4; ++i) {
      const int idx = tid + i * 256;
      const int r   = idx >> 3;
      const int c4  = (idx & 7) * 4;
      float4 a = *(const float4*)(x    + (size_t)(m0 + r) * C_ + k0 + c4);
      float4 b = *(const float4*)(Wqkv + (size_t)(n0 + r) * C_ + k0 + c4);
      uint2 pa, pb;
      pa.x = (uint32_t)f2bf(a.x) | ((uint32_t)f2bf(a.y) << 16);
      pa.y = (uint32_t)f2bf(a.z) | ((uint32_t)f2bf(a.w) << 16);
      pb.x = (uint32_t)f2bf(b.x) | ((uint32_t)f2bf(b.y) << 16);
      pb.y = (uint32_t)f2bf(b.z) | ((uint32_t)f2bf(b.w) << 16);
      *(uint2*)(As + r * LDSS + c4) = pa;
      *(uint2*)(Bs + r * LDSS + c4) = pb;
    }
    __syncthreads();
    v16bf af[4], bfr[2];
#pragma unroll
    for (int i = 0; i < 4; ++i) af[i]  = frag_a(As, LDSS, wm + i * 16, 0, lane);
#pragma unroll
    for (int j = 0; j < 2; ++j) bfr[j] = frag_b(Bs, LDSS, wn + j * 16, 0, lane);
#pragma unroll
    for (int i = 0; i < 4; ++i)
#pragma unroll
      for (int j = 0; j < 2; ++j)
        acc[i][j] = wmma_bf16(af[i], bfr[j], acc[i][j]);
    __syncthreads();
  }

  const int hf = lane >> 4;
#pragma unroll
  for (int i = 0; i < 4; ++i) {
#pragma unroll
    for (int j = 0; j < 2; ++j) {
      const int n    = n0 + wn + j * 16 + (lane & 15);
      const int mg0  = m0 + wm + i * 16 + 8 * hf;
      const float bias = bqkv[n];
      const int sec = n >> 11;
      const int nn  = n & (C_ - 1);
      const int h   = nn >> 7;
      const int d   = nn & (HD_ - 1);
#pragma unroll
      for (int e = 0; e < 8; ++e) {
        const int m = mg0 + e;
        const float val = acc[i][j][e] + bias;
        const int b = m >> 11;
        const int t = m & (T_ - 1);
        const size_t hidx = (((size_t)(b * NH_ + h)) * T_ + t) * HD_ + d;
        if (sec == 0) {
          qb[hidx] = f2bf(val);
        } else if (sec == 1) {
          outk[hidx] = val;
          kbuf[hidx] = f2bf(val);
        } else {
          outv[hidx] = val;
          vbuf[hidx] = f2bf(val);
        }
      }
    }
  }
}

// ---------------------------------------------------------------------------
// Kernel 2: flash attention, causal. Block = (b*h, 128 q rows), 8 waves.
// K tile staged by TDM (or async-to-LDS); V transposed at staging.
// ---------------------------------------------------------------------------
__launch_bounds__(256)
__global__ void attn_kernel(const unsigned short* __restrict__ qb,
                            const unsigned short* __restrict__ kbuf,
                            const unsigned short* __restrict__ vbuf,
                            unsigned short* __restrict__ ob) {
  constexpr int LK = 136;
  constexpr int LV = 72;
  constexpr int LP = 72;
  __shared__ __align__(16) unsigned short Ks[64 * LK];
  __shared__ __align__(16) unsigned short Vt[HD_ * LV];
  __shared__ __align__(16) unsigned short Ps[8 * 16 * LP];

  const int tid  = threadIdx.x;
  const int lane = tid & 31;
  const int wave = tid >> 5;
  const int hf   = lane >> 4;
  const int bh = blockIdx.x;
  const int q0 = blockIdx.y * 128;
  const size_t base = (size_t)bh * T_ * HD_;
  const int qrow0 = q0 + wave * 16;

  v16bf qf[4];
  {
    const int m = lane & 15;
    const unsigned short* qp = qb + base + (size_t)(qrow0 + m) * HD_;
#pragma unroll
    for (int s = 0; s < 4; ++s) {
      FragBF f;
      f.u4[0] = *(const uint4*)(qp + s * 32 + hf * 8);
      f.u4[1] = *(const uint4*)(qp + s * 32 + 16 + hf * 8);
      qf[s] = f.v;
    }
  }

  v8f o[8];
#pragma unroll
  for (int ct = 0; ct < 8; ++ct) o[ct] = zero8();
  float mrow[8], lrow[8];
#pragma unroll
  for (int e = 0; e < 8; ++e) { mrow[e] = -1e30f; lrow[e] = 0.f; }

  const float scale = 0.08838834764831845f;  // 1/sqrt(128)
  const int nkb = (q0 >> 6) + 2;

  for (int kblk = 0; kblk < nkb; ++kblk) {
    const int kt0 = kblk * 64;
#if HAS_TDM
    if (wave == 0) tdm_load_k_tile(kbuf + base + (size_t)kt0 * HD_, Ks);
#endif
#pragma unroll
    for (int i = 0; i < 4; ++i) {
      const int idx = tid + i * 256;
      const int r   = idx >> 4;
      const int c8  = (idx & 15) * 8;
#if !HAS_TDM
      copy16_g2l(kbuf + base + (size_t)(kt0 + r) * HD_ + c8, Ks + r * LK + c8);
#endif
      uint4 vv = *(const uint4*)(vbuf + base + (size_t)(kt0 + r) * HD_ + c8);
      const unsigned short* ve = (const unsigned short*)&vv;
#pragma unroll
      for (int j = 0; j < 8; ++j) Vt[(c8 + j) * LV + r] = ve[j];
    }
#if HAS_TDM
    if (wave == 0) wait_tensor();
#else
    wait_async_lds();
#endif
    __syncthreads();

    if (kt0 <= qrow0) {
      v8f sa[4];
#pragma unroll
      for (int nt = 0; nt < 4; ++nt) {
        sa[nt] = zero8();
#pragma unroll
        for (int s = 0; s < 4; ++s) {
          v16bf kf = frag_b(Ks, LK, nt * 16, s * 32, lane);
          sa[nt] = wmma_bf16(qf[s], kf, sa[nt]);
        }
      }
#pragma unroll
      for (int e = 0; e < 8; ++e) {
        const int qr = qrow0 + 8 * hf + e;
        float sv[4];
        float vmax = -1e30f;
#pragma unroll
        for (int nt = 0; nt < 4; ++nt) {
          const int kc = kt0 + nt * 16 + (lane & 15);
          float s = sa[nt][e] * scale;
          s = (kc <= qr) ? s : -1e30f;
          sv[nt] = s;
          vmax = fmaxf(vmax, s);
        }
#pragma unroll
        for (int off = 1; off < 16; off <<= 1)
          vmax = fmaxf(vmax, __shfl_xor(vmax, off, 32));
        const float mnew  = fmaxf(mrow[e], vmax);
        const float alpha = __expf(mrow[e] - mnew);
        float psum = 0.f;
#pragma unroll
        for (int nt = 0; nt < 4; ++nt) { sv[nt] = __expf(sv[nt] - mnew); psum += sv[nt]; }
#pragma unroll
        for (int off = 1; off < 16; off <<= 1)
          psum += __shfl_xor(psum, off, 32);
        lrow[e] = lrow[e] * alpha + psum;
        mrow[e] = mnew;
#pragma unroll
        for (int ct = 0; ct < 8; ++ct) o[ct][e] *= alpha;
        unsigned short* pw = Ps + wave * 16 * LP + (8 * hf + e) * LP + (lane & 15);
        pw[0]  = f2bf(sv[0]);
        pw[16] = f2bf(sv[1]);
        pw[32] = f2bf(sv[2]);
        pw[48] = f2bf(sv[3]);
      }
      const unsigned short* pb = Ps + wave * 16 * LP;
#pragma unroll
      for (int ks = 0; ks < 2; ++ks) {
        v16bf pf = frag_a(pb, LP, 0, ks * 32, lane);
#pragma unroll
        for (int ct = 0; ct < 8; ++ct) {
          v16bf vf = frag_b(Vt, LV, ct * 16, ks * 32, lane);
          o[ct] = wmma_bf16(pf, vf, o[ct]);
        }
      }
    }
    __syncthreads();
  }

  const int b = bh >> 4;
  const int h = bh & 15;
#pragma unroll
  for (int ct = 0; ct < 8; ++ct) {
    const int d = h * HD_ + ct * 16 + (lane & 15);
#pragma unroll
    for (int e = 0; e < 8; ++e) {
      const int t = qrow0 + 8 * hf + e;
      const float val = o[ct][e] / lrow[e];
      ob[((size_t)(b * T_ + t)) * C_ + d] = f2bf(val);
    }
  }
}

// ---------------------------------------------------------------------------
// Kernel 3: y = attn_out @ Wout^T + bout
// ---------------------------------------------------------------------------
__launch_bounds__(256)
__global__ void out_gemm_kernel(const unsigned short* __restrict__ ob,
                                const float* __restrict__ Wout,
                                const float* __restrict__ bout,
                                float* __restrict__ y) {
  constexpr int LDSS = 40;
  __shared__ __align__(16) unsigned short As[128 * LDSS];
  __shared__ __align__(16) unsigned short Bs[128 * LDSS];
  const int tid  = threadIdx.x;
  const int lane = tid & 31;
  const int wave = tid >> 5;
  const int m0 = blockIdx.x * 128;
  const int n0 = blockIdx.y * 128;
  const int wm = (wave >> 2) * 64;
  const int wn = (wave & 3) * 32;

  v8f acc[4][2];
#pragma unroll
  for (int i = 0; i < 4; ++i)
#pragma unroll
    for (int j = 0; j < 2; ++j) acc[i][j] = zero8();

  for (int k0 = 0; k0 < C_; k0 += 32) {
    // A already bf16: async direct-to-LDS copy
#pragma unroll
    for (int i = 0; i < 2; ++i) {
      const int idx = tid + i * 256;
      const int r   = idx >> 2;
      const int c8  = (idx & 3) * 8;
      copy16_g2l(ob + (size_t)(m0 + r) * C_ + k0 + c8, As + r * LDSS + c8);
    }
    // B: fp32 -> bf16 convert on stage
#pragma unroll
    for (int i = 0; i < 4; ++i) {
      const int idx = tid + i * 256;
      const int r   = idx >> 3;
      const int c4  = (idx & 7) * 4;
      float4 b = *(const float4*)(Wout + (size_t)(n0 + r) * C_ + k0 + c4);
      uint2 pb;
      pb.x = (uint32_t)f2bf(b.x) | ((uint32_t)f2bf(b.y) << 16);
      pb.y = (uint32_t)f2bf(b.z) | ((uint32_t)f2bf(b.w) << 16);
      *(uint2*)(Bs + r * LDSS + c4) = pb;
    }
    wait_async_lds();
    __syncthreads();
    v16bf af[4], bfr[2];
#pragma unroll
    for (int i = 0; i < 4; ++i) af[i]  = frag_a(As, LDSS, wm + i * 16, 0, lane);
#pragma unroll
    for (int j = 0; j < 2; ++j) bfr[j] = frag_b(Bs, LDSS, wn + j * 16, 0, lane);
#pragma unroll
    for (int i = 0; i < 4; ++i)
#pragma unroll
      for (int j = 0; j < 2; ++j)
        acc[i][j] = wmma_bf16(af[i], bfr[j], acc[i][j]);
    __syncthreads();
  }

  const int hf = lane >> 4;
#pragma unroll
  for (int i = 0; i < 4; ++i) {
#pragma unroll
    for (int j = 0; j < 2; ++j) {
      const int n   = n0 + wn + j * 16 + (lane & 15);
      const int mg0 = m0 + wm + i * 16 + 8 * hf;
      const float bias = bout[n];
#pragma unroll
      for (int e = 0; e < 8; ++e) {
        const int m = mg0 + e;
        y[(size_t)m * C_ + n] = acc[i][j][e] + bias;
      }
    }
  }
}

// ---------------------------------------------------------------------------
extern "C" void kernel_launch(void* const* d_in, const int* in_sizes, int n_in,
                              void* d_out, int out_size, void* d_ws, size_t ws_size,
                              hipStream_t stream) {
  (void)in_sizes; (void)n_in; (void)out_size; (void)ws_size;
  const float* x    = (const float*)d_in[0];
  const float* Wqkv = (const float*)d_in[1];
  const float* bqkv = (const float*)d_in[2];
  const float* Wout = (const float*)d_in[3];
  const float* bout = (const float*)d_in[4];

  const size_t NTC = (size_t)B_ * T_ * C_;
  float* y    = (float*)d_out;
  float* outk = y + NTC;
  float* outv = outk + NTC;

  unsigned short* qb   = (unsigned short*)d_ws;
  unsigned short* kbuf = qb + NTC;
  unsigned short* vbuf = kbuf + NTC;
  unsigned short* ob   = vbuf + NTC;

  qkv_gemm_kernel<<<dim3(64, 48), 256, 0, stream>>>(x, Wqkv, bqkv, outk, outv,
                                                    qb, kbuf, vbuf);
  attn_kernel<<<dim3(64, 16), 256, 0, stream>>>(qb, kbuf, vbuf, ob);
  out_gemm_kernel<<<dim3(64, 16), 256, 0, stream>>>(ob, Wout, bout, y);
}
